// SelfAttention_26474178412982
// MI455X (gfx1250) — compile-verified, matrix operand order checked by reading, not run on hardware
//
#include <hip/hip_runtime.h>
#include <hip/hip_bf16.h>

// ---------------------------------------------------------------------------
// Self-attention forward for MI455X (gfx1250), wave32 + WMMA bf16.
//   B=4, S=2048, D=H=1024
//   q = x@Wq^T + bq ; k = ... ; v = ...
//   scores = q@k^T * 1/32 ; attn = softmax(scores) ; out = attn@v
//   d_out = [ weighted_sum (4*2048*1024 f32) | attn (4*2048*2048 f32) ]
//
// GEMMs: 256 threads = 8 waves, 128x128 block tile, 64x32 wave tile.
// K pipeline: double-buffered LDS staging via async global->LDS (ASYNCcnt);
// v_wmma_f32_16x16x32_bf16 compute from ds_load_b128 fragments.
// ---------------------------------------------------------------------------

typedef __bf16 bf16;
typedef __bf16 v16bf __attribute__((ext_vector_type(16)));
typedef __bf16 v8bf  __attribute__((ext_vector_type(8)));
typedef __bf16 v4bf  __attribute__((ext_vector_type(4)));
typedef float  v8f   __attribute__((ext_vector_type(8)));
typedef int    v4i   __attribute__((ext_vector_type(4)));

#define BQ 4
#define SEQ 2048
#define DIM 1024
#define MROWS (BQ * SEQ)   // 8192

#define LPAD 40            // padded LDS row stride (elems): 80B = 20 banks -> conflict-free

// Epilogue modes
#define OUT_BF16   0   // bf16 row-major + bias        (q, k)
#define OUT_BF16_T 1   // bf16 transposed + bias       (v -> vT[b][H][S])
#define OUT_F32    2   // f32 row-major, scaled        (scores, attn@v)

#if defined(__has_builtin)
#if __has_builtin(__builtin_amdgcn_global_load_async_to_lds_b128)
#define HAVE_ASYNC_LDS 1
#endif
#endif
#ifndef HAVE_ASYNC_LDS
#define HAVE_ASYNC_LDS 0
#endif

__device__ __forceinline__ void wait_async_zero() {
#if HAVE_ASYNC_LDS
#if __has_builtin(__builtin_amdgcn_s_wait_asynccnt)
    __builtin_amdgcn_s_wait_asynccnt(0);
#else
    asm volatile("s_wait_asynccnt 0x0" ::: "memory");
#endif
#endif
}

// Copy one 16-byte chunk global -> LDS (async engine, ASYNCcnt-tracked).
__device__ __forceinline__ void copy_b128(const bf16* __restrict__ gp, bf16* lp) {
#if HAVE_ASYNC_LDS
    __builtin_amdgcn_global_load_async_to_lds_b128(
        (__attribute__((address_space(1))) v4i*)(unsigned long long)(const void*)gp,
        (__attribute__((address_space(3))) v4i*)(unsigned int)(unsigned long long)(void*)lp,
        0, 0);
#else
    *(v8bf*)lp = *(const v8bf*)gp;
#endif
}

// ---------------------------------------------------------------------------
// fp32 -> bf16 elementwise convert (vectorized x4)
// ---------------------------------------------------------------------------
__global__ __launch_bounds__(256) void f32_to_bf16_kernel(const float* __restrict__ src,
                                                          bf16* __restrict__ dst, int n4) {
    int i = blockIdx.x * blockDim.x + threadIdx.x;
    if (i < n4) {
        float4 f = ((const float4*)src)[i];
        v4bf o;
        o.x = (bf16)f.x; o.y = (bf16)f.y; o.z = (bf16)f.z; o.w = (bf16)f.w;
        *(v4bf*)(dst + 4 * (size_t)i) = o;
    }
}

// ---------------------------------------------------------------------------
// Stage one 128x32 bf16 tile into LDS (rows padded to LPAD).
// 256 threads x 2 chunks of 8 elems = 4096 elems. Fully coalesced 16B/lane.
// ---------------------------------------------------------------------------
__device__ __forceinline__ void stage_tile(const bf16* __restrict__ gsrc, int gld,
                                           int row0, int kk, bf16* lds, int tid) {
#pragma unroll
    for (int h = 0; h < 2; ++h) {
        const int c   = tid + h * 256;     // 0..511
        const int row = c >> 2;            // 0..127
        const int col = (c & 3) * 8;       // 0,8,16,24
        copy_b128(gsrc + (size_t)(row0 + row) * gld + kk + col,
                  lds + row * LPAD + col);
    }
}

// ---------------------------------------------------------------------------
// Build one 16x32 bf16 fragment from an LDS tile (A layout; B-frag == A of B^T).
// Lane L: row M = L&15; lanes<16 take K {0..7,16..23}, lanes>=16 {8..15,24..31}.
// ---------------------------------------------------------------------------
__device__ __forceinline__ v16bf lds_frag(const bf16* lds, int row0, int lane) {
    const int m     = lane & 15;
    const int khalf = (lane >> 4) << 3;          // 0 or 8
    const bf16* p   = lds + (row0 + m) * LPAD + khalf;
    v8bf lo = *(const v8bf*)(p);                 // ds_load_b128
    v8bf hi = *(const v8bf*)(p + 16);            // ds_load_b128
    return __builtin_shufflevector(lo, hi, 0, 1, 2, 3, 4, 5, 6, 7,
                                           8, 9, 10, 11, 12, 13, 14, 15);
}

// ---------------------------------------------------------------------------
// WMMA GEMM: C[M,N] = A[M,K] * Bt[N,K]^T  (Bt stored row-major = B^T)
// ---------------------------------------------------------------------------
template <int MODE>
__global__ __launch_bounds__(256) void gemm_wmma(const bf16* __restrict__ A,
                                                 const bf16* __restrict__ Bt,
                                                 const float* __restrict__ bias,
                                                 void* __restrict__ outv,
                                                 int K, int lda, int ldb, int ldc,
                                                 long aBatch, long bBatch, long oBatch,
                                                 float scale) {
    __shared__ bf16 lA[2][128 * LPAD];
    __shared__ bf16 lB[2][128 * LPAD];

    const int tid  = threadIdx.x;
    const int lane = tid & 31;
    const int wv   = tid >> 5;
    const int bm   = blockIdx.x * 128;
    const int bn   = blockIdx.y * 128;
    const int wmr  = (wv >> 2) * 64;   // wave M origin, block-relative (4 x 16)
    const int wnr  = (wv & 3) * 32;    // wave N origin, block-relative (2 x 16)

    const bf16* Ab = A  + (size_t)aBatch * blockIdx.z;
    const bf16* Bb = Bt + (size_t)bBatch * blockIdx.z;

    v8f acc[4][2] = {};

    // Prologue: stage first K-tile
    stage_tile(Ab, lda, bm, 0, lA[0], tid);
    stage_tile(Bb, ldb, bn, 0, lB[0], tid);
    wait_async_zero();
    __syncthreads();

    for (int kk = 0; kk < K; kk += 32) {
        const int cur = (kk >> 5) & 1;
        const bool more = (kk + 32) < K;

        // Prefetch next K-tile into the other buffer (overlaps with compute)
        if (more) {
            stage_tile(Ab, lda, bm, kk + 32, lA[cur ^ 1], tid);
            stage_tile(Bb, ldb, bn, kk + 32, lB[cur ^ 1], tid);
        }

        v16bf af[4], bfm[2];
#pragma unroll
        for (int i = 0; i < 4; ++i)
            af[i] = lds_frag(lA[cur], wmr + i * 16, lane);
#pragma unroll
        for (int j = 0; j < 2; ++j)
            bfm[j] = lds_frag(lB[cur], wnr + j * 16, lane);
#pragma unroll
        for (int i = 0; i < 4; ++i)
#pragma unroll
            for (int j = 0; j < 2; ++j)
                acc[i][j] = __builtin_amdgcn_wmma_f32_16x16x32_bf16(
                    false, af[i], false, bfm[j], (short)0, acc[i][j], false, false);

        if (more) wait_async_zero();   // own async fills done before barrier
        __syncthreads();               // all waves: buffer ready / reads finished
    }

    // C/D layout: VGPR r -> M = r + (lane>=16 ? 8 : 0);  N = lane & 15
    const int rsel = (lane >> 4) << 3;
    const int csel = lane & 15;
    const int wm0  = bm + wmr;
    const int wn0  = bn + wnr;

    if (MODE == OUT_F32) {
        float* O = (float*)outv + (size_t)oBatch * blockIdx.z;
#pragma unroll
        for (int i = 0; i < 4; ++i)
#pragma unroll
            for (int j = 0; j < 2; ++j) {
                const int col = wn0 + j * 16 + csel;
#pragma unroll
                for (int r = 0; r < 8; ++r) {
                    const int row = wm0 + i * 16 + r + rsel;
                    O[(size_t)row * ldc + col] = acc[i][j][r] * scale;
                }
            }
    } else if (MODE == OUT_BF16) {
        bf16* O = (bf16*)outv;
#pragma unroll
        for (int j = 0; j < 2; ++j) {
            const int col = wn0 + j * 16 + csel;
            const float bj = bias[col];
#pragma unroll
            for (int i = 0; i < 4; ++i)
#pragma unroll
                for (int r = 0; r < 8; ++r) {
                    const int row = wm0 + i * 16 + r + rsel;
                    O[(size_t)row * ldc + col] = (bf16)(acc[i][j][r] + bj);
                }
        }
    } else { // OUT_BF16_T: out laid out [b][N(=H)][S], row = b*S + s
        bf16* O = (bf16*)outv;
#pragma unroll
        for (int j = 0; j < 2; ++j) {
            const int col = wn0 + j * 16 + csel;
            const float bj = bias[col];
#pragma unroll
            for (int i = 0; i < 4; ++i)
#pragma unroll
                for (int r = 0; r < 8; ++r) {
                    const int row = wm0 + i * 16 + r + rsel;
                    const int b   = row >> 11;        // / SEQ
                    const int s   = row & 2047;       // % SEQ
                    O[((size_t)b * DIM + col) * SEQ + s] = (bf16)(acc[i][j][r] + bj);
                }
        }
    }
}

// ---------------------------------------------------------------------------
// Row softmax over 2048 f32 (in place) + bf16 probability copy for the PV GEMM.
// One 256-thread block per row; 8 elements/thread.
// ---------------------------------------------------------------------------
__global__ __launch_bounds__(256) void softmax_rows(float* __restrict__ attn,
                                                    bf16* __restrict__ probs) {
    __shared__ float red[256];
    const size_t row = blockIdx.x;
    float* p = attn  + row * (size_t)SEQ;
    bf16*  q = probs + row * (size_t)SEQ;
    const int t = threadIdx.x;

    float4 a = ((const float4*)p)[t];
    float4 b = ((const float4*)p)[t + 256];

    float m = fmaxf(fmaxf(fmaxf(a.x, a.y), fmaxf(a.z, a.w)),
                    fmaxf(fmaxf(b.x, b.y), fmaxf(b.z, b.w)));
    red[t] = m;
    __syncthreads();
    for (int s = 128; s > 0; s >>= 1) {
        if (t < s) red[t] = fmaxf(red[t], red[t + s]);
        __syncthreads();
    }
    m = red[0];
    __syncthreads();

    a.x = __expf(a.x - m); a.y = __expf(a.y - m);
    a.z = __expf(a.z - m); a.w = __expf(a.w - m);
    b.x = __expf(b.x - m); b.y = __expf(b.y - m);
    b.z = __expf(b.z - m); b.w = __expf(b.w - m);

    red[t] = a.x + a.y + a.z + a.w + b.x + b.y + b.z + b.w;
    __syncthreads();
    for (int s = 128; s > 0; s >>= 1) {
        if (t < s) red[t] += red[t + s];
        __syncthreads();
    }
    const float inv = 1.0f / red[0];

    a.x *= inv; a.y *= inv; a.z *= inv; a.w *= inv;
    b.x *= inv; b.y *= inv; b.z *= inv; b.w *= inv;

    ((float4*)p)[t]       = a;
    ((float4*)p)[t + 256] = b;

    v4bf qa, qb;
    qa.x = (bf16)a.x; qa.y = (bf16)a.y; qa.z = (bf16)a.z; qa.w = (bf16)a.w;
    qb.x = (bf16)b.x; qb.y = (bf16)b.y; qb.z = (bf16)b.z; qb.w = (bf16)b.w;
    *(v4bf*)(q + 4 * t)         = qa;
    *(v4bf*)(q + 4 * (t + 256)) = qb;
}

// ---------------------------------------------------------------------------
extern "C" void kernel_launch(void* const* d_in, const int* in_sizes, int n_in,
                              void* d_out, int out_size, void* d_ws, size_t ws_size,
                              hipStream_t stream) {
    const float* x  = (const float*)d_in[0];
    const float* Wq = (const float*)d_in[1];
    const float* bq = (const float*)d_in[2];
    const float* Wk = (const float*)d_in[3];
    const float* bk = (const float*)d_in[4];
    const float* Wv = (const float*)d_in[5];
    const float* bv = (const float*)d_in[6];

    float* out  = (float*)d_out;                      // [4,2048,1024] weighted sum
    float* attn = out + (size_t)MROWS * DIM;          // [4,2048,2048] attention weights

    // workspace layout (bf16 staging, all 256B aligned)
    char* ws = (char*)d_ws;
    bf16* xb  = (bf16*)(ws);                          // 16 MB  x
    bf16* Wqb = (bf16*)(ws + 16777216);               //  2 MB
    bf16* Wkb = (bf16*)(ws + 18874368);               //  2 MB
    bf16* Wvb = (bf16*)(ws + 20971520);               //  2 MB
    bf16* qb  = (bf16*)(ws + 23068672);               // 16 MB  q
    bf16* kb  = (bf16*)(ws + 39845888);               // 16 MB  k
    bf16* vTb = (bf16*)(ws + 56623104);               // 16 MB  v^T [b][H][S]
    bf16* pb  = (bf16*)(ws + 73400320);               // 32 MB  probs bf16

    const float scale = 0.03125f;                     // 1/sqrt(1024)

    // 1) fp32 -> bf16 conversions
    {
        int n4 = MROWS * DIM / 4;                     // 2M
        f32_to_bf16_kernel<<<(n4 + 255) / 256, 256, 0, stream>>>(x, xb, n4);
        int w4 = DIM * DIM / 4;                       // 256K
        f32_to_bf16_kernel<<<(w4 + 255) / 256, 256, 0, stream>>>(Wq, Wqb, w4);
        f32_to_bf16_kernel<<<(w4 + 255) / 256, 256, 0, stream>>>(Wk, Wkb, w4);
        f32_to_bf16_kernel<<<(w4 + 255) / 256, 256, 0, stream>>>(Wv, Wvb, w4);
    }

    dim3 blk(256);

    // 2) QKV projections: [8192,1024] = xb[8192,1024] @ W^T ; W row-major == B^T
    gemm_wmma<OUT_BF16><<<dim3(64, 8, 1), blk, 0, stream>>>(
        xb, Wqb, bq, qb, DIM, DIM, DIM, DIM, 0, 0, 0, 1.0f);
    gemm_wmma<OUT_BF16><<<dim3(64, 8, 1), blk, 0, stream>>>(
        xb, Wkb, bk, kb, DIM, DIM, DIM, DIM, 0, 0, 0, 1.0f);
    gemm_wmma<OUT_BF16_T><<<dim3(64, 8, 1), blk, 0, stream>>>(
        xb, Wvb, bv, vTb, DIM, DIM, DIM, DIM, 0, 0, 0, 1.0f);

    // 3) scores = q @ k^T * scale  (per batch; k row-major == B^T)
    gemm_wmma<OUT_F32><<<dim3(16, 16, BQ), blk, 0, stream>>>(
        qb, kb, nullptr, attn, DIM, DIM, DIM, SEQ,
        (long)SEQ * DIM, (long)SEQ * DIM, (long)SEQ * SEQ, scale);

    // 4) softmax rows (f32 in place into d_out, bf16 copy for PV)
    softmax_rows<<<MROWS, 256, 0, stream>>>(attn, pb);

    // 5) out = attn @ v  (A = probs bf16 [S,S]; B^T = vT rows [H,S])
    gemm_wmma<OUT_F32><<<dim3(16, 8, BQ), blk, 0, stream>>>(
        pb, vTb, nullptr, out, SEQ, SEQ, SEQ, DIM,
        (long)SEQ * SEQ, (long)DIM * SEQ, (long)SEQ * DIM, 1.0f);
}
